// Forest_13030930776174
// MI455X (gfx1250) — matrix operand order repaired and checked by reading, not compiled, not run on hardware
//
#include <hip/hip_runtime.h>

typedef __attribute__((ext_vector_type(16))) __bf16 v16bf;
typedef __attribute__((ext_vector_type(8)))  __bf16 v8bf;
typedef __attribute__((ext_vector_type(8)))  float  v8f;

#define B_BATCH 4096
#define F_DIM   1024
#define T_TREES 64
#define S_NODES 511
#define L_LEAF  512
#define C_CLS   101
#define C_PAD   128

// ---------------------------------------------------------------- convert
__global__ void forest_f32_to_bf16(const float* __restrict__ src,
                                   __bf16* __restrict__ dst, int n) {
  int i = blockIdx.x * blockDim.x + threadIdx.x;
  if (i < n) dst[i] = (__bf16)src[i];
}

// ---------------------------------------------------------------- h = sigmoid(x @ W^T), WMMA bf16
// block = 128 threads (4 waves); block tile 16 x 256; wave tile 16 x 64 (4 N-tiles, A reused)
__global__ __launch_bounds__(128)
void forest_gemm_sigmoid(const __bf16* __restrict__ xb,
                         const __bf16* __restrict__ wb,
                         float* __restrict__ h) {
  const int lane = threadIdx.x & 31;
  const int wave = threadIdx.x >> 5;
  const int row0 = blockIdx.x * 16;                  // batch tile
  const int col0 = blockIdx.y * 256 + wave * 64;     // 4 feature tiles per wave
  const int m    = lane & 15;
  const int hiA  = (lane >> 4) << 4;                 // 0 or 16 (A: K split 0..15 / 16..31)
  const int hiB  = (lane >> 4) << 4;                 // 0 or 16 (B: K split 0..15 / 16..31)

  const __bf16* arow = xb + (size_t)(row0 + m) * F_DIM;
  const __bf16* brow0 = wb + (size_t)(col0 +  0 + m) * F_DIM; // W[f][k] row-major == B col f
  const __bf16* brow1 = wb + (size_t)(col0 + 16 + m) * F_DIM;
  const __bf16* brow2 = wb + (size_t)(col0 + 32 + m) * F_DIM;
  const __bf16* brow3 = wb + (size_t)(col0 + 48 + m) * F_DIM;

  v8f acc0 = {}, acc1 = {}, acc2 = {}, acc3 = {};
  #pragma unroll 2
  for (int k0 = 0; k0 < F_DIM; k0 += 32) {
    const __bf16* ap = arow + k0 + hiA;
    v8bf alo = *(const v8bf*)ap;         // K = k0+hiA+0..7
    v8bf ahi = *(const v8bf*)(ap + 8);   // K = k0+hiA+8..15
    v16bf a;
    #pragma unroll
    for (int i = 0; i < 8; ++i) { a[i] = alo[i]; a[8 + i] = ahi[i]; }
    v16bf b0 = *(const v16bf*)(brow0 + k0 + hiB);
    v16bf b1 = *(const v16bf*)(brow1 + k0 + hiB);
    v16bf b2 = *(const v16bf*)(brow2 + k0 + hiB);
    v16bf b3 = *(const v16bf*)(brow3 + k0 + hiB);
    acc0 = __builtin_amdgcn_wmma_f32_16x16x32_bf16(false, a, false, b0, (short)0, acc0, false, false);
    acc1 = __builtin_amdgcn_wmma_f32_16x16x32_bf16(false, a, false, b1, (short)0, acc1, false, false);
    acc2 = __builtin_amdgcn_wmma_f32_16x16x32_bf16(false, a, false, b2, (short)0, acc2, false, false);
    acc3 = __builtin_amdgcn_wmma_f32_16x16x32_bf16(false, a, false, b3, (short)0, acc3, false, false);
  }
  // sigmoid via fast rcp (v_rcp_f32), store f32
  const int mout = (lane >> 4) << 3;
  #pragma unroll
  for (int r = 0; r < 8; ++r) {
    size_t rowoff = (size_t)(row0 + r + mout) * F_DIM + (col0 + m);
    h[rowoff +  0] = __builtin_amdgcn_rcpf(1.0f + __expf(-acc0[r]));
    h[rowoff + 16] = __builtin_amdgcn_rcpf(1.0f + __expf(-acc1[r]));
    h[rowoff + 32] = __builtin_amdgcn_rcpf(1.0f + __expf(-acc2[r]));
    h[rowoff + 48] = __builtin_amdgcn_rcpf(1.0f + __expf(-acc3[r]));
  }
}

// ---------------------------------------------------------------- softmax(pi) -> P2[t][c][l] bf16, padded to C_PAD
__global__ void forest_softmax_pack(const float* __restrict__ pi,
                                    __bf16* __restrict__ p2) {
  int i = blockIdx.x * blockDim.x + threadIdx.x;  // one thread per (t,l)
  if (i >= T_TREES * L_LEAF) return;
  int t = i >> 9;          // / L_LEAF
  int l = i & (L_LEAF - 1);
  const float* row = pi + ((size_t)t * L_LEAF + l) * C_CLS;
  float mx = row[0];
  for (int c = 1; c < C_CLS; ++c) mx = fmaxf(mx, row[c]);
  float sum = 0.f;
  for (int c = 0; c < C_CLS; ++c) sum += __expf(row[c] - mx);
  float inv = __builtin_amdgcn_rcpf(sum);
  __bf16* dst = p2 + (size_t)t * C_PAD * L_LEAF;
  for (int c = 0; c < C_CLS; ++c) dst[(size_t)c * L_LEAF + l] = (__bf16)(__expf(row[c] - mx) * inv);
  for (int c = C_CLS; c < C_PAD; ++c) dst[(size_t)c * L_LEAF + l] = (__bf16)0.f;
}

// ---------------------------------------------------------------- fused tree expansion + einsum GEMM
// LDS: hT 16x1024 f32 (64KB) | muT 16x512 bf16 (16KB) | fmC 512 int (2KB)
#define SM_HT_BYTES  (16 * F_DIM * 4)
#define SM_MU_BYTES  (16 * L_LEAF * 2)
#define SM_FM_BYTES  (L_LEAF * 4)
#define SM_TOTAL     (SM_HT_BYTES + SM_MU_BYTES + SM_FM_BYTES)

__global__ __launch_bounds__(128)
void forest_tree_einsum(const float* __restrict__ h,
                        const int* __restrict__ fm,
                        const __bf16* __restrict__ p2,
                        float* __restrict__ out) {
  extern __shared__ char smem[];
  float*  hT  = (float*)smem;
  __bf16* muT = (__bf16*)(smem + SM_HT_BYTES);
  int*    fmC = (int*)(smem + SM_HT_BYTES + SM_MU_BYTES);

  const int tid  = threadIdx.x;
  const int lane = tid & 31;
  const int wave = tid >> 5;
  const int b0   = blockIdx.x * 16;

  // stage 16 sigmoid rows
  for (int i = tid; i < 16 * F_DIM; i += 128)
    hT[i] = h[(size_t)(b0 + (i >> 10)) * F_DIM + (i & (F_DIM - 1))];

  const int r  = tid >> 3;   // batch row 0..15 for tree expansion
  const int g  = tid & 7;    // leaf group (64 leaves each)
  const int mA  = lane & 15;
  const int hiA = (lane >> 4) << 4;
  const int hiB = (lane >> 4) << 4;
  const int n0a = wave * 32;        // this wave's two N tiles
  const int n0b = wave * 32 + 16;

  v8f acc0 = {};
  v8f acc1 = {};

  for (int t = 0; t < T_TREES; ++t) {
    for (int s = tid; s < S_NODES; s += 128) fmC[s] = fm[s * T_TREES + t];
    __syncthreads();

    // depth-9 tree: levels 0-2 hoisted prefix, levels 3-8 iterative doubling in regs
    const float* hrow = hT + r * F_DIM;
    float d0 = hrow[fmC[0]];
    float d1 = hrow[fmC[1 + (g >> 2)]];
    float d2 = hrow[fmC[3 + (g >> 1)]];
    float pref = ((g & 4) ? (1.f - d0) : d0)
               * ((g & 2) ? (1.f - d1) : d1)
               * ((g & 1) ? (1.f - d2) : d2);
    float accL[64];
    accL[0] = pref;
    #pragma unroll
    for (int lvl = 3; lvl < 9; ++lvl) {
      const int cnt = 1 << (lvl - 3);
      const int lo  = (1 << lvl) - 1 + (g << (lvl - 3));
      #pragma unroll
      for (int i2 = cnt - 1; i2 >= 0; --i2) {
        float d = hrow[fmC[lo + i2]];
        float mu = accL[i2];
        accL[2 * i2]     = mu * d;
        accL[2 * i2 + 1] = mu * (1.f - d);
      }
    }
    __bf16* mrow = muT + r * L_LEAF + g * 64;
    #pragma unroll
    for (int j = 0; j < 64; ++j) mrow[j] = (__bf16)accL[j];
    __syncthreads();

    // mu_t (16x512) @ P2_t (512x128): 16 k-steps, 2 N tiles per wave
    const __bf16* pt = p2 + (size_t)t * C_PAD * L_LEAF;
    #pragma unroll 2
    for (int k0 = 0; k0 < L_LEAF; k0 += 32) {
      const __bf16* ap = muT + mA * L_LEAF + k0 + hiA;
      v8bf alo = *(const v8bf*)ap;
      v8bf ahi = *(const v8bf*)(ap + 8);
      v16bf a;
      #pragma unroll
      for (int i2 = 0; i2 < 8; ++i2) { a[i2] = alo[i2]; a[8 + i2] = ahi[i2]; }
      v16bf bb0 = *(const v16bf*)(pt + (size_t)(n0a + mA) * L_LEAF + k0 + hiB);
      v16bf bb1 = *(const v16bf*)(pt + (size_t)(n0b + mA) * L_LEAF + k0 + hiB);
      acc0 = __builtin_amdgcn_wmma_f32_16x16x32_bf16(false, a, false, bb0,
                                                     (short)0, acc0, false, false);
      acc1 = __builtin_amdgcn_wmma_f32_16x16x32_bf16(false, a, false, bb1,
                                                     (short)0, acc1, false, false);
    }
    __syncthreads();
  }

  // D layout: VGPR rr -> (M = rr + 8*(lane>=16), N = lane&15); scale by 1/T
  const int mout = (lane >> 4) << 3;
  const int cA = n0a + (lane & 15);
  const int cB = n0b + (lane & 15);
  #pragma unroll
  for (int rr = 0; rr < 8; ++rr) {
    int brow = b0 + rr + mout;
    if (cA < C_CLS) out[(size_t)brow * C_CLS + cA] = acc0[rr] * (1.f / 64.f);
    if (cB < C_CLS) out[(size_t)brow * C_CLS + cB] = acc1[rr] * (1.f / 64.f);
  }
}

// ---------------------------------------------------------------- launch
extern "C" void kernel_launch(void* const* d_in, const int* in_sizes, int n_in,
                              void* d_out, int out_size, void* d_ws, size_t ws_size,
                              hipStream_t stream) {
  const float* x  = (const float*)d_in[0];
  const float* W  = (const float*)d_in[1];
  const float* pi = (const float*)d_in[2];
  const int*   fm = (const int*)d_in[3];
  float* out = (float*)d_out;
  char* ws = (char*)d_ws;

  // workspace layout (bytes)
  const size_t XB_BYTES = (size_t)B_BATCH * F_DIM * 2;   // 8 MB bf16 x
  const size_t WB_BYTES = (size_t)F_DIM * F_DIM * 2;     // 2 MB bf16 W
  const size_t H_BYTES  = (size_t)B_BATCH * F_DIM * 4;   // 16 MB f32 sigmoid(xW^T)
  __bf16* xb = (__bf16*)ws;
  __bf16* wb = (__bf16*)(ws + XB_BYTES);
  float*  h  = (float*)(ws + XB_BYTES + WB_BYTES);
  __bf16* p2 = (__bf16*)(ws + XB_BYTES + WB_BYTES + H_BYTES); // 8 MB packed softmax(pi)

  int nx = B_BATCH * F_DIM;
  int nw = F_DIM * F_DIM;
  forest_f32_to_bf16<<<(nx + 255) / 256, 256, 0, stream>>>(x, xb, nx);
  forest_f32_to_bf16<<<(nw + 255) / 256, 256, 0, stream>>>(W, wb, nw);

  forest_gemm_sigmoid<<<dim3(B_BATCH / 16, F_DIM / 256), 128, 0, stream>>>(xb, wb, h);

  forest_softmax_pack<<<(T_TREES * L_LEAF + 255) / 256, 256, 0, stream>>>(pi, p2);

  forest_tree_einsum<<<B_BATCH / 16, 128, SM_TOTAL, stream>>>(h, fm, p2, out);
}
